// LSTMBlock_30142080483674
// MI455X (gfx1250) — matrix operand order repaired and checked by reading, not compiled, not run on hardware
//
#include <hip/hip_runtime.h>
#include <hip/hip_bf16.h>

typedef __attribute__((ext_vector_type(16))) _Float16 v16h;
typedef __attribute__((ext_vector_type(4)))  _Float16 v4h;
typedef __attribute__((ext_vector_type(8)))  float    v8f;

#define HD 128          // H_DIM == C_DIM
#define KD 256          // 2*H
#define PACK1_HALVES (32*8*32*16)   // 32 tiles x 8 ktiles x 32 lanes x 16 f16
#define PACK2_HALVES (8*4*32*16)    // 8 tiles x 4 ktiles x 32 lanes x 16 f16

#define LOG2E 1.4426950408889634f

// sigmoid: v_mul + v_exp_f32 + v_add + v_rcp_f32  (no IEEE divide expansion)
__device__ __forceinline__ float fast_sigmoid(float x) {
    return __builtin_amdgcn_rcpf(1.0f + __builtin_amdgcn_exp2f(-LOG2E * x));
}
// tanh(x) = 2*sigmoid(2x) - 1 ; saturates cleanly (exp2->inf => rcp->0)
__device__ __forceinline__ float fast_tanh(float x) {
    float s = __builtin_amdgcn_rcpf(1.0f + __builtin_amdgcn_exp2f(-2.0f * LOG2E * x));
    return fmaf(2.0f, s, -1.0f);
}

// ---------------------------------------------------------------------------
// Pack W_gates (Wf|Wi|Wc|Wo, each [256,128] row-major) and Wch ([128,128])
// into f16 WMMA B-fragment order:
//   pack[tile][kt][lane][j], value = W[K, n] with
//   K = kt*32 + (j<8 ? 8*(lane>>4)+j : 16+8*(lane>>4)+j-8),  n = tile_col*16 + (lane&15)
// ---------------------------------------------------------------------------
__global__ void pack_weights(const float* __restrict__ Wf, const float* __restrict__ Wi,
                             const float* __restrict__ Wc, const float* __restrict__ Wo,
                             const float* __restrict__ Wch,
                             _Float16* __restrict__ pack1, _Float16* __restrict__ pack2) {
    int idx = blockIdx.x * blockDim.x + threadIdx.x;
    if (idx < PACK1_HALVES) {
        int j    = idx & 15;
        int lane = (idx >> 4) & 31;
        int kt   = (idx >> 9) & 7;
        int nt   = idx >> 12;            // 0..31 : nt = g*8 + w
        int g = nt >> 3, w = nt & 7;
        int K = kt * 32 + ((j < 8) ? 8 * (lane >> 4) + j : 16 + 8 * (lane >> 4) + (j - 8));
        int n = w * 16 + (lane & 15);
        const float* W = (g == 0) ? Wf : (g == 1) ? Wi : (g == 2) ? Wc : Wo;
        pack1[idx] = (_Float16)W[K * HD + n];
    } else if (idx < PACK1_HALVES + PACK2_HALVES) {
        int p = idx - PACK1_HALVES;
        int j    = p & 15;
        int lane = (p >> 4) & 31;
        int kt   = (p >> 9) & 3;
        int nt   = p >> 11;              // 0..7
        int K = kt * 32 + ((j < 8) ? 8 * (lane >> 4) + j : 16 + 8 * (lane >> 4) + (j - 8));
        int n = nt * 16 + (lane & 15);
        pack2[p] = (_Float16)Wch[K * HD + n];
    }
}

// ---------------------------------------------------------------------------
// Main fused LSTM kernel. Block = 256 threads (8 waves), 64 rows per block.
// Wave w owns gate-column slice [w*16, w*16+16) for all four gates, so the
// elementwise gate math stays entirely in registers (fragments line up).
// kt loops kept rolled (unroll 1) to bound live B-fragments -> <256 VGPRs.
// ---------------------------------------------------------------------------
__global__ __launch_bounds__(256)
void lstm_main(const float* __restrict__ x, const float* __restrict__ h,
               const float* __restrict__ c,
               const float* __restrict__ bf, const float* __restrict__ bi,
               const float* __restrict__ bc, const float* __restrict__ bo,
               const float* __restrict__ bch,
               const _Float16* __restrict__ pack1, const _Float16* __restrict__ pack2,
               float* __restrict__ h_out, float* __restrict__ c_out) {
    // A1 fragments: 4 mtiles x 8 ktiles x (32 lanes x 16 f16)  = 32 KB
    // A2 fragments: 4 mtiles x 4 ktiles x (32 lanes x 16 f16)  = 16 KB
    __shared__ __align__(32) _Float16 ldsA1[4 * 8 * 32 * 16];
    __shared__ __align__(32) _Float16 ldsA2[4 * 4 * 32 * 16];

    const int tid  = threadIdx.x;
    const int lane = tid & 31;
    const int wv   = tid >> 5;           // wave id 0..7 = gate-column slice
    const int row0 = blockIdx.x * 64;

    // ---- Phase 1: stage concat(x,h) rows [row0,row0+64) as f16 A-fragments.
    // 4 consecutive K (4-aligned) land in 4 consecutive fragment slots, so we
    // load float4 from global and store v4h (8B) to LDS.
    #pragma unroll 4
    for (int it = 0; it < 16; ++it) {
        int e4 = tid + it * 256;         // 0..4095 chunks of 4 elements
        int m  = e4 >> 6;                // 0..63 (64 chunks per row)
        int k4 = (e4 & 63) * 4;          // 0,4,...,252
        const float* src = (k4 < HD) ? (x + (row0 + m) * HD + k4)
                                     : (h + (row0 + m) * HD + (k4 - HD));
        float4 v = *(const float4*)src;
        int mt = m >> 4, ml = m & 15;
        int kt = k4 >> 5, kk = k4 & 31;
        int hi = (kk >> 3) & 1;
        int j  = (kk & 7) + ((kk & 16) ? 8 : 0);
        v4h pk = {(_Float16)v.x, (_Float16)v.y, (_Float16)v.z, (_Float16)v.w};
        *(v4h*)(ldsA1 + ((mt * 8 + kt) * 32 + (ml + 16 * hi)) * 16 + j) = pk;
    }
    __syncthreads();

    // ---- Phase 2: GEMM1 — gates[m, g*128 + wv*16 + n] for 4 mtiles
    const int nl = lane & 15;
    v8f acc[4][4];
    {
        float bv0 = bf[wv * 16 + nl];
        float bv1 = bi[wv * 16 + nl];
        float bv2 = bc[wv * 16 + nl];
        float bv3 = bo[wv * 16 + nl];
        #pragma unroll
        for (int mt = 0; mt < 4; ++mt) {
            acc[0][mt] = (v8f){bv0, bv0, bv0, bv0, bv0, bv0, bv0, bv0};
            acc[1][mt] = (v8f){bv1, bv1, bv1, bv1, bv1, bv1, bv1, bv1};
            acc[2][mt] = (v8f){bv2, bv2, bv2, bv2, bv2, bv2, bv2, bv2};
            acc[3][mt] = (v8f){bv3, bv3, bv3, bv3, bv3, bv3, bv3, bv3};
        }
    }
    #pragma unroll 1
    for (int kt = 0; kt < 8; ++kt) {
        v16h bfr[4];
        #pragma unroll
        for (int g = 0; g < 4; ++g)
            bfr[g] = *(const v16h*)(pack1 + (((g * 8 + wv) * 8 + kt) * 32 + lane) * 16);
        #pragma unroll
        for (int mt = 0; mt < 4; ++mt) {
            v16h afr = *(const v16h*)(ldsA1 + ((mt * 8 + kt) * 32 + lane) * 16);
            #pragma unroll
            for (int g = 0; g < 4; ++g)
                acc[g][mt] = __builtin_amdgcn_wmma_f32_16x16x32_f16(
                    false, afr, false, bfr[g], (short)0, acc[g][mt], false, false);
        }
    }

    // ---- Phase 3: gate math in registers; scatter c_new to LDS as A2 frags
    const int hi  = lane >> 4;           // D layout: M = v + 8*hi
    const int col = wv * 16 + nl;        // 0..127
    const int kt2 = col >> 5;
    const int kk2 = col & 31;
    const int hi2 = (kk2 >> 3) & 1;
    const int j2  = (kk2 & 7) + ((kk2 & 16) ? 8 : 0);
    #pragma unroll
    for (int mt = 0; mt < 4; ++mt) {
        #pragma unroll
        for (int e = 0; e < 8; ++e) {
            int   mloc = e + 8 * hi;                 // row within 16-row tile
            int   row  = row0 + mt * 16 + mloc;
            float fe = fast_sigmoid(acc[0][mt][e]);
            float ie = fast_sigmoid(acc[1][mt][e]);
            float ce = fast_tanh  (acc[2][mt][e]);
            float oe = fast_sigmoid(acc[3][mt][e]);
            float cn = fmaf(fe, c[row * HD + col], ie * ce);
            c_out[row * HD + col] = cn;
            acc[3][mt][e] = oe;                      // keep o for phase 4
            int lane2 = mloc + 16 * hi2;
            ldsA2[((mt * 4 + kt2) * 32 + lane2) * 16 + j2] = (_Float16)cn;
        }
    }
    __syncthreads();

    // ---- Phase 4: GEMM2 — h_new = o * tanh(c_new @ Wch + bch)
    float bv2 = bch[col];
    v8f acc2[4];
    #pragma unroll
    for (int mt = 0; mt < 4; ++mt)
        acc2[mt] = (v8f){bv2, bv2, bv2, bv2, bv2, bv2, bv2, bv2};
    #pragma unroll 1
    for (int kt = 0; kt < 4; ++kt) {
        v16h bfr = *(const v16h*)(pack2 + ((wv * 4 + kt) * 32 + lane) * 16);
        #pragma unroll
        for (int mt = 0; mt < 4; ++mt) {
            v16h afr = *(const v16h*)(ldsA2 + ((mt * 4 + kt) * 32 + lane) * 16);
            acc2[mt] = __builtin_amdgcn_wmma_f32_16x16x32_f16(
                false, afr, false, bfr, (short)0, acc2[mt], false, false);
        }
    }
    #pragma unroll
    for (int mt = 0; mt < 4; ++mt) {
        #pragma unroll
        for (int e = 0; e < 8; ++e) {
            int row = row0 + mt * 16 + e + 8 * hi;
            h_out[row * HD + col] = acc[3][mt][e] * fast_tanh(acc2[mt][e]);
        }
    }
}

extern "C" void kernel_launch(void* const* d_in, const int* in_sizes, int n_in,
                              void* d_out, int out_size, void* d_ws, size_t ws_size,
                              hipStream_t stream) {
    const float* x   = (const float*)d_in[0];
    const float* h   = (const float*)d_in[1];
    const float* c   = (const float*)d_in[2];
    const float* Wf  = (const float*)d_in[3];
    const float* bf  = (const float*)d_in[4];
    const float* Wi  = (const float*)d_in[5];
    const float* bi  = (const float*)d_in[6];
    const float* Wc  = (const float*)d_in[7];
    const float* bc  = (const float*)d_in[8];
    const float* Wo  = (const float*)d_in[9];
    const float* bo  = (const float*)d_in[10];
    const float* Wch = (const float*)d_in[11];
    const float* bch = (const float*)d_in[12];

    const int Btot = in_sizes[0] / HD;       // 262144

    _Float16* pack1 = (_Float16*)d_ws;
    _Float16* pack2 = pack1 + PACK1_HALVES;

    const int packTotal = PACK1_HALVES + PACK2_HALVES;   // 147456
    pack_weights<<<(packTotal + 255) / 256, 256, 0, stream>>>(
        Wf, Wi, Wc, Wo, Wch, pack1, pack2);

    float* h_out = (float*)d_out;
    float* c_out = h_out + (size_t)Btot * HD;

    lstm_main<<<Btot / 64, 256, 0, stream>>>(
        x, h, c, bf, bi, bc, bo, bch, pack1, pack2, h_out, c_out);
}
